// MoELayer_51307679318138
// MI455X (gfx1250) — matrix operand order repaired and checked by reading, not compiled, not run on hardware
//
#include <hip/hip_runtime.h>
#include <hip/hip_bf16.h>
#include <stdint.h>

// Problem dims (B*T tokens)
#define NTOK 4096
#define DDIM 1024
#define HDIM 4096
#define ODIM 1024
#define NEXP 8
#define NENT (NTOK * 2)   // total routed assignments (top-2)
#define MTILE 32          // tokens per GEMM block

typedef __attribute__((ext_vector_type(16))) __bf16        v16bf;
typedef __attribute__((ext_vector_type(8)))  float         v8f;
typedef __attribute__((ext_vector_type(8)))  unsigned int  v8u;
typedef __attribute__((ext_vector_type(4)))  unsigned int  u32x4;
typedef __attribute__((ext_vector_type(4)))  unsigned short u16x4;

__device__ __forceinline__ unsigned short f2bf(float f) {
  unsigned int u = __float_as_uint(f);
  unsigned int r = u + 0x7FFFu + ((u >> 16) & 1u);   // round-to-nearest-even
  return (unsigned short)(r >> 16);
}

__device__ __forceinline__ v16bf as_bf16(v8u v) { return __builtin_bit_cast(v16bf, v); }

// Async copy of one 16B chunk global->LDS (per lane), tracked by ASYNCcnt.
__device__ __forceinline__ void async_ldst128(unsigned lds_byte_addr, const void* gptr) {
  unsigned long long ga = (unsigned long long)(uintptr_t)gptr;
  asm volatile("global_load_async_to_lds_b128 %0, %1, off"
               :: "v"(lds_byte_addr), "v"(ga) : "memory");
}
__device__ __forceinline__ void async_wait0() {
  asm volatile("s_wait_asynccnt 0" ::: "memory");
}

// ---------------------------------------------------------------- x -> bf16
__global__ __launch_bounds__(256) void cvt_x_k(const float* __restrict__ x,
                                               unsigned short* __restrict__ xb) {
  int i = (blockIdx.x * 256 + threadIdx.x) * 4;
  float4 v = *(const float4*)(x + i);
  u16x4 o;
  o[0] = f2bf(v.x); o[1] = f2bf(v.y); o[2] = f2bf(v.z); o[3] = f2bf(v.w);
  *(u16x4*)(xb + i) = o;
}

// ------------------------------------- per-expert transpose + f32->bf16
// in: [R][C] f32 row-major, out: [C][R] bf16. blockIdx.z = expert.
__global__ __launch_bounds__(256) void transpose_cvt_k(const float* __restrict__ in,
                                                       unsigned short* __restrict__ out,
                                                       int R, int C) {
  __shared__ unsigned short t[32][33];
  size_t mat = (size_t)blockIdx.z * (size_t)R * (size_t)C;
  const float* ip = in + mat;
  unsigned short* op = out + mat;
  int c0 = blockIdx.x * 32, r0 = blockIdx.y * 32;
  int tx = threadIdx.x, ty = threadIdx.y;          // block (32,8)
  for (int i = ty; i < 32; i += 8)
    t[i][tx] = f2bf(ip[(size_t)(r0 + i) * C + (c0 + tx)]);
  __syncthreads();
  for (int i = ty; i < 32; i += 8)
    op[(size_t)(c0 + i) * R + (r0 + tx)] = t[tx][i];
}

// ---------------------------------------------------------------- gating
// one wave32 per token: scores = softmax(x @ Wg + bg); top-2; L1-normalize.
__global__ __launch_bounds__(256) void gating_k(const float* __restrict__ x,
                                                const float* __restrict__ Wg,
                                                const float* __restrict__ bg,
                                                int* __restrict__ cnt,
                                                int* __restrict__ tinfo_i,
                                                float* __restrict__ tinfo_w) {
  int tok  = (blockIdx.x * 256 + threadIdx.x) >> 5;
  int lane = threadIdx.x & 31;
  const float* xr = x + (size_t)tok * DDIM;
  float acc[NEXP];
#pragma unroll
  for (int e = 0; e < NEXP; e++) acc[e] = 0.f;
  for (int d = lane; d < DDIM; d += 32) {
    float xv = xr[d];
    const float* wr = Wg + d * NEXP;
#pragma unroll
    for (int e = 0; e < NEXP; e++) acc[e] += xv * wr[e];
  }
#pragma unroll
  for (int off = 16; off > 0; off >>= 1) {
#pragma unroll
    for (int e = 0; e < NEXP; e++) acc[e] += __shfl_xor(acc[e], off, 32);
  }
  if (lane == 0) {
    float mx = -3.4e38f;
#pragma unroll
    for (int e = 0; e < NEXP; e++) { acc[e] += bg[e]; mx = fmaxf(mx, acc[e]); }
    float p[NEXP], sum = 0.f;
#pragma unroll
    for (int e = 0; e < NEXP; e++) { p[e] = __expf(acc[e] - mx); sum += p[e]; }
    int i0 = 0;
#pragma unroll
    for (int e = 1; e < NEXP; e++) if (p[e] > p[i0]) i0 = e;
    int i1 = (i0 == 0) ? 1 : 0;
#pragma unroll
    for (int e = 0; e < NEXP; e++) if (e != i0 && p[e] > p[i1]) i1 = e;
    float inv = 1.f / sum;
    float g0 = p[i0] * inv, g1 = p[i1] * inv;
    float l1 = fmaxf(g0 + g1, 1e-12f);
    tinfo_i[2 * tok] = i0;  tinfo_i[2 * tok + 1] = i1;
    tinfo_w[2 * tok] = g0 / l1;  tinfo_w[2 * tok + 1] = g1 / l1;
    atomicAdd(&cnt[i0], 1);  atomicAdd(&cnt[i1], 1);
  }
}

// ------------------------------------------------- exclusive scan over E=8
__global__ void scan_k(const int* __restrict__ cnt, int* __restrict__ offs,
                       int* __restrict__ cursor) {
  if (threadIdx.x == 0) {
    int acc = 0;
    for (int e = 0; e < NEXP; e++) { offs[e] = acc; acc += cnt[e]; cursor[e] = 0; }
  }
}

// --------------------------------------- scatter tokens into expert lists
__global__ __launch_bounds__(256) void scatter_k(const int* __restrict__ tinfo_i,
                                                 const float* __restrict__ tinfo_w,
                                                 const int* __restrict__ offs,
                                                 int* __restrict__ cursor,
                                                 int* __restrict__ entry_tok,
                                                 int* __restrict__ entry_slot,
                                                 float* __restrict__ entry_w) {
  int t = blockIdx.x * 256 + threadIdx.x;
  if (t >= NTOK) return;
#pragma unroll
  for (int s = 0; s < 2; s++) {
    int e = tinfo_i[2 * t + s];
    int pos = offs[e] + atomicAdd(&cursor[e], 1);
    entry_tok[pos]  = t;
    entry_slot[pos] = s;
    entry_w[pos]    = tinfo_w[2 * t + s];
  }
}

// --------------------------------------------- GEMM1: hidden = relu(x@W1+b1)
// grid (NTOK/32, E), 256 threads = 8 waves. W1t: [E][H][D] bf16.
// A tile 32x1024 bf16 = 64 KB dynamic LDS, filled via async-to-LDS.
__global__ __launch_bounds__(256) void gemm1_k(const unsigned short* __restrict__ xb,
                                               const unsigned short* __restrict__ W1t,
                                               const float* __restrict__ b1,
                                               const int* __restrict__ cnt,
                                               const int* __restrict__ offs,
                                               const int* __restrict__ entry_tok,
                                               unsigned short* __restrict__ hidden) {
  const int e = blockIdx.y;
  const int count = cnt[e];
  const int mtile = blockIdx.x;
  if (mtile * MTILE >= count) return;
  const int base = offs[e] + mtile * MTILE;
  const int rows_valid = min(MTILE, count - mtile * MTILE);

  extern __shared__ __align__(16) unsigned short lA[];   // [32][DDIM]
  __shared__ int toks[MTILE];
  const int tid = threadIdx.x;
  const int lane = tid & 31, wv = tid >> 5;
  if (tid < MTILE) toks[tid] = entry_tok[base + min(tid, rows_valid - 1)];
  __syncthreads();

  // async fill: wave wv stages rows 4wv..4wv+3; 2048B/row = 128 x 16B chunks
  const unsigned ldsbase = (unsigned)(uintptr_t)lA;
#pragma unroll
  for (int rr = 0; rr < 4; rr++) {
    const int r = wv * 4 + rr;
    const unsigned short* rowp = xb + (size_t)toks[r] * DDIM;
#pragma unroll
    for (int i = 0; i < 4; i++) {
      const int chunk = i * 32 + lane;               // 512B contiguous per wave op
      async_ldst128(ldsbase + (unsigned)(r * DDIM + chunk * 8) * 2, rowp + chunk * 8);
    }
  }
  async_wait0();
  __syncthreads();

  const int m16 = lane & 15, sel = lane >> 4;
  const unsigned short* Wbase = W1t + (size_t)e * HDIM * DDIM;
  const unsigned short* arow0 = lA + m16 * DDIM + sel * 8;          // M rows 0..15
  const unsigned short* arow1 = lA + (16 + m16) * DDIM + sel * 8;   // M rows 16..31

  for (int j = 0; j < 32; j++) {                 // 256 N-tiles / 8 waves
    const int ntile = j * 8 + wv;
    const int col = ntile * 16 + m16;
    const unsigned short* bptr = Wbase + (size_t)col * DDIM + sel * 8;
    v8f acc0 = {0.f, 0.f, 0.f, 0.f, 0.f, 0.f, 0.f, 0.f};
    v8f acc1 = {0.f, 0.f, 0.f, 0.f, 0.f, 0.f, 0.f, 0.f};
#pragma unroll 4
    for (int k = 0; k < DDIM; k += 32) {
      u32x4 b0 = *(const u32x4*)(bptr + k);       // shared by both M-subtiles
      u32x4 b1v = *(const u32x4*)(bptr + k + 16);
      v8u bv;
      bv[0] = b0[0]; bv[1] = b0[1]; bv[2] = b0[2]; bv[3] = b0[3];
      bv[4] = b1v[0]; bv[5] = b1v[1]; bv[6] = b1v[2]; bv[7] = b1v[3];
      u32x4 a0 = *(const u32x4*)(arow0 + k);
      u32x4 a1 = *(const u32x4*)(arow0 + k + 16);
      v8u av0;
      av0[0] = a0[0]; av0[1] = a0[1]; av0[2] = a0[2]; av0[3] = a0[3];
      av0[4] = a1[0]; av0[5] = a1[1]; av0[6] = a1[2]; av0[7] = a1[3];
      u32x4 a2 = *(const u32x4*)(arow1 + k);
      u32x4 a3 = *(const u32x4*)(arow1 + k + 16);
      v8u av1;
      av1[0] = a2[0]; av1[1] = a2[1]; av1[2] = a2[2]; av1[3] = a2[3];
      av1[4] = a3[0]; av1[5] = a3[1]; av1[6] = a3[2]; av1[7] = a3[3];
      acc0 = __builtin_amdgcn_wmma_f32_16x16x32_bf16(
          false, as_bf16(av0), false, as_bf16(bv), (short)0, acc0, false, false);
      acc1 = __builtin_amdgcn_wmma_f32_16x16x32_bf16(
          false, as_bf16(av1), false, as_bf16(bv), (short)0, acc1, false, false);
    }
    const float bias = b1[e * HDIM + col];
#pragma unroll
    for (int i = 0; i < 8; i++) {
      const int m = i + sel * 8;
      if (m < rows_valid)
        hidden[(size_t)(base + m) * HDIM + col] = f2bf(fmaxf(acc0[i] + bias, 0.f));
      if (m + 16 < rows_valid)
        hidden[(size_t)(base + m + 16) * HDIM + col] = f2bf(fmaxf(acc1[i] + bias, 0.f));
    }
  }
}

// ------------------------------- GEMM2: contrib = gate * (hidden@W2 + b2)
// grid (NTOK/32, E). W2t: [E][O][H] bf16. Whole K=4096 A-tile resident:
// 32x4096 bf16 = 256 KB dynamic LDS (within 320 KB WGP LDS).
__global__ __launch_bounds__(256) void gemm2_k(const unsigned short* __restrict__ hidden,
                                               const unsigned short* __restrict__ W2t,
                                               const float* __restrict__ b2,
                                               const int* __restrict__ cnt,
                                               const int* __restrict__ offs,
                                               const int* __restrict__ entry_tok,
                                               const int* __restrict__ entry_slot,
                                               const float* __restrict__ entry_w,
                                               float* __restrict__ contrib) {
  const int e = blockIdx.y;
  const int count = cnt[e];
  const int mtile = blockIdx.x;
  if (mtile * MTILE >= count) return;
  const int base = offs[e] + mtile * MTILE;
  const int rows_valid = min(MTILE, count - mtile * MTILE);

  extern __shared__ __align__(16) unsigned short lA[];   // [32][HDIM]
  __shared__ int toks[MTILE];
  __shared__ int slots[MTILE];
  __shared__ float gw[MTILE];
  const int tid = threadIdx.x;
  const int lane = tid & 31, wv = tid >> 5;
  if (tid < MTILE) {
    int r = min(tid, rows_valid - 1);
    toks[tid]  = entry_tok[base + r];
    slots[tid] = entry_slot[base + r];
    gw[tid]    = entry_w[base + r];
  }

  // async fill: wave wv stages rows 4wv..4wv+3; 8192B/row = 512 x 16B chunks
  const unsigned ldsbase = (unsigned)(uintptr_t)lA;
#pragma unroll
  for (int rr = 0; rr < 4; rr++) {
    const int r = wv * 4 + rr;
    const unsigned short* rowp = hidden + (size_t)(base + r) * HDIM;
    for (int i = 0; i < 16; i++) {
      const int chunk = i * 32 + lane;
      async_ldst128(ldsbase + (unsigned)(r * HDIM + chunk * 8) * 2, rowp + chunk * 8);
    }
  }
  async_wait0();
  __syncthreads();

  const int m16 = lane & 15, sel = lane >> 4;
  const unsigned short* Wbase = W2t + (size_t)e * ODIM * HDIM;
  const unsigned short* arow0 = lA + m16 * HDIM + sel * 8;
  const unsigned short* arow1 = lA + (16 + m16) * HDIM + sel * 8;

  for (int j = 0; j < 8; j++) {                  // 64 N-tiles / 8 waves
    const int ntile = j * 8 + wv;
    const int col = ntile * 16 + m16;
    const unsigned short* bptr = Wbase + (size_t)col * HDIM + sel * 8;
    v8f acc0 = {0.f, 0.f, 0.f, 0.f, 0.f, 0.f, 0.f, 0.f};
    v8f acc1 = {0.f, 0.f, 0.f, 0.f, 0.f, 0.f, 0.f, 0.f};
#pragma unroll 4
    for (int k = 0; k < HDIM; k += 32) {
      u32x4 b0 = *(const u32x4*)(bptr + k);
      u32x4 b1v = *(const u32x4*)(bptr + k + 16);
      v8u bv;
      bv[0] = b0[0]; bv[1] = b0[1]; bv[2] = b0[2]; bv[3] = b0[3];
      bv[4] = b1v[0]; bv[5] = b1v[1]; bv[6] = b1v[2]; bv[7] = b1v[3];
      u32x4 a0 = *(const u32x4*)(arow0 + k);
      u32x4 a1 = *(const u32x4*)(arow0 + k + 16);
      v8u av0;
      av0[0] = a0[0]; av0[1] = a0[1]; av0[2] = a0[2]; av0[3] = a0[3];
      av0[4] = a1[0]; av0[5] = a1[1]; av0[6] = a1[2]; av0[7] = a1[3];
      u32x4 a2 = *(const u32x4*)(arow1 + k);
      u32x4 a3 = *(const u32x4*)(arow1 + k + 16);
      v8u av1;
      av1[0] = a2[0]; av1[1] = a2[1]; av1[2] = a2[2]; av1[3] = a2[3];
      av1[4] = a3[0]; av1[5] = a3[1]; av1[6] = a3[2]; av1[7] = a3[3];
      acc0 = __builtin_amdgcn_wmma_f32_16x16x32_bf16(
          false, as_bf16(av0), false, as_bf16(bv), (short)0, acc0, false, false);
      acc1 = __builtin_amdgcn_wmma_f32_16x16x32_bf16(
          false, as_bf16(av1), false, as_bf16(bv), (short)0, acc1, false, false);
    }
    const float bias = b2[e * ODIM + col];
#pragma unroll
    for (int i = 0; i < 8; i++) {
      const int m = i + sel * 8;
      if (m < rows_valid) {
        const float v = (acc0[i] + bias) * gw[m];
        contrib[((size_t)slots[m] * NTOK + toks[m]) * ODIM + col] = v;
      }
      if (m + 16 < rows_valid) {
        const float v = (acc1[i] + bias) * gw[m + 16];
        contrib[((size_t)slots[m + 16] * NTOK + toks[m + 16]) * ODIM + col] = v;
      }
    }
  }
}

// -------------------------------------------------- out = contrib0+contrib1
__global__ __launch_bounds__(256) void combine_k(const float* __restrict__ contrib,
                                                 float* __restrict__ out) {
  int i = (blockIdx.x * 256 + threadIdx.x) * 4;
  float4 a = *(const float4*)(contrib + i);
  float4 b = *(const float4*)(contrib + (size_t)NTOK * ODIM + i);
  float4 r;
  r.x = a.x + b.x; r.y = a.y + b.y; r.z = a.z + b.z; r.w = a.w + b.w;
  *(float4*)(out + i) = r;
}

extern "C" void kernel_launch(void* const* d_in, const int* in_sizes, int n_in,
                              void* d_out, int out_size, void* d_ws, size_t ws_size,
                              hipStream_t stream) {
  (void)in_sizes; (void)n_in; (void)out_size; (void)ws_size;
  const float* x  = (const float*)d_in[0];
  const float* Wg = (const float*)d_in[1];
  const float* bg = (const float*)d_in[2];
  const float* W1 = (const float*)d_in[3];
  const float* b1 = (const float*)d_in[4];
  const float* W2 = (const float*)d_in[5];
  const float* b2 = (const float*)d_in[6];
  float* out = (float*)d_out;

  char* p = (char*)d_ws;
  auto alloc = [&](size_t bytes) -> void* {
    void* r = (void*)p;
    p += (bytes + 255) & ~(size_t)255;
    return r;
  };
  unsigned short* xb  = (unsigned short*)alloc((size_t)NTOK * DDIM * 2);
  unsigned short* W1t = (unsigned short*)alloc((size_t)NEXP * DDIM * HDIM * 2);
  unsigned short* W2t = (unsigned short*)alloc((size_t)NEXP * HDIM * ODIM * 2);
  unsigned short* hid = (unsigned short*)alloc((size_t)(NENT + MTILE) * HDIM * 2);
  float* contrib      = (float*)alloc((size_t)2 * NTOK * ODIM * 4);
  int*   tinfo_i      = (int*)alloc((size_t)NTOK * 2 * 4);
  float* tinfo_w      = (float*)alloc((size_t)NTOK * 2 * 4);
  int*   entry_tok    = (int*)alloc((size_t)NENT * 4);
  int*   entry_slot   = (int*)alloc((size_t)NENT * 4);
  float* entry_w      = (float*)alloc((size_t)NENT * 4);
  int*   cnt          = (int*)alloc(NEXP * 4);
  int*   offs         = (int*)alloc(NEXP * 4);
  int*   cursor       = (int*)alloc(NEXP * 4);

  // Allow >default dynamic LDS (CDNA5 WGP: up to 320 KB per workgroup).
  hipFuncSetAttribute((const void*)gemm1_k, hipFuncAttributeMaxDynamicSharedMemorySize,
                      MTILE * DDIM * 2);
  hipFuncSetAttribute((const void*)gemm2_k, hipFuncAttributeMaxDynamicSharedMemorySize,
                      MTILE * HDIM * 2);

  hipMemsetAsync(cnt, 0, NEXP * sizeof(int), stream);
  cvt_x_k<<<(NTOK * DDIM) / (256 * 4), 256, 0, stream>>>(x, xb);
  transpose_cvt_k<<<dim3(HDIM / 32, DDIM / 32, NEXP), dim3(32, 8), 0, stream>>>(W1, W1t, DDIM, HDIM);
  transpose_cvt_k<<<dim3(ODIM / 32, HDIM / 32, NEXP), dim3(32, 8), 0, stream>>>(W2, W2t, HDIM, ODIM);
  gating_k<<<NTOK / 8, 256, 0, stream>>>(x, Wg, bg, cnt, tinfo_i, tinfo_w);
  scan_k<<<1, 32, 0, stream>>>(cnt, offs, cursor);
  scatter_k<<<NTOK / 256, 256, 0, stream>>>(tinfo_i, tinfo_w, offs, cursor,
                                            entry_tok, entry_slot, entry_w);
  gemm1_k<<<dim3(NTOK / MTILE, NEXP), 256, MTILE * DDIM * 2, stream>>>(
      xb, W1t, b1, cnt, offs, entry_tok, hid);
  gemm2_k<<<dim3(NTOK / MTILE, NEXP), 256, MTILE * HDIM * 2, stream>>>(
      hid, W2t, b2, cnt, offs, entry_tok, entry_slot, entry_w, contrib);
  combine_k<<<(NTOK * ODIM) / (256 * 4), 256, 0, stream>>>(contrib, out);
}
